// dcell_5437428597470
// MI455X (gfx1250) — compile-verified, matrix operand order checked by reading, not run on hardware
//
#include <hip/hip_runtime.h>
#include <cmath>

// ---------------- problem constants ----------------
#define BATCH 4096
#define NGENE 3000
#define HID   6
#define GPT   30
#define T0N   500
#define T1N   50
#define NTERM 551              // T0+T1+1
#define NCOL  16530            // NTERM*GPT  (N of big GEMM)
#define KDIM  3000             // K of big GEMM
#define KP    3008             // K padded to multiple of 32
#define NP    16896            // N padded to multiple of 128 (132*128)
#define OUTW  3857             // 551 aux + 3306 nn
#define NNOFF 551
#define EPSBN 1e-5f

typedef __bf16 bf16_t;
typedef __attribute__((ext_vector_type(16))) __bf16 v16bf;
typedef __attribute__((ext_vector_type(8)))  __bf16 v8bf;
typedef __attribute__((ext_vector_type(8)))  float  v8f;

// ---------------- fp32 -> split bf16 (hi + lo), RNE ----------------
__device__ __forceinline__ unsigned short f2bf(float f) {
    unsigned int u = __float_as_uint(f);
    u += 0x7FFFu + ((u >> 16) & 1u);          // round to nearest even
    return (unsigned short)(u >> 16);
}
__device__ __forceinline__ float bf2f(unsigned short h) {
    return __uint_as_float(((unsigned int)h) << 16);
}

// src[rows, cols] fp32 -> hi/lo[rows_pad, cols_pad] bf16, zero padded
__global__ void cvt_split_kernel(const float* __restrict__ src, int rows, int cols,
                                 unsigned short* __restrict__ hi,
                                 unsigned short* __restrict__ lo,
                                 int rows_pad, int cols_pad) {
    long long idx = (long long)blockIdx.x * blockDim.x + threadIdx.x;
    long long total = (long long)rows_pad * cols_pad;
    if (idx >= total) return;
    int r = (int)(idx / cols_pad);
    int c = (int)(idx % cols_pad);
    float v = (r < rows && c < cols) ? src[(long long)r * cols + c] : 0.0f;
    unsigned short h = f2bf(v);
    hi[idx] = h;
    lo[idx] = f2bf(v - bf2f(h));
}

// ---------------- big GEMM: dg = x @ Wdg^T + bdg via split-bf16 WMMA ----------------
// ISA 16-bit A-frag layout (16x32): lane<16 holds K 0..7 / 16..23 of row (lane&15),
// lane>=16 holds K 8..15 / 24..31. B (K x N) is loaded with the identical pattern
// since Wdg is stored [N, K] row-major (== column-major B).
__device__ __forceinline__ v16bf load_frag(const unsigned short* __restrict__ base,
                                           int row, int k, int laneHi) {
    const unsigned short* p = base + (long long)row * KP + k + (laneHi ? 8 : 0);
    union { v16bf f; v8bf h[2]; } u;
    u.h[0] = *(const v8bf*)(p);
    u.h[1] = *(const v8bf*)(p + 16);
    return u.f;
}

__global__ void __launch_bounds__(256)
gemm_dg_kernel(const unsigned short* __restrict__ Ahi,
               const unsigned short* __restrict__ Alo,
               const unsigned short* __restrict__ Whi,
               const unsigned short* __restrict__ Wlo,
               const float* __restrict__ bdg,
               float* __restrict__ dg) {
    const int lane   = threadIdx.x & 31;
    const int wave   = threadIdx.x >> 5;       // 0..7
    const int laneHi = lane >> 4;              // 0/1
    const int laneLo = lane & 15;
    const int waveM  = wave & 3;               // 4 waves along M
    const int waveN  = wave >> 2;              // 2 waves along N
    const int mb = blockIdx.y * 128 + waveM * 32;   // 2 M-tiles of 16
    const int nb = blockIdx.x * 128 + waveN * 64;   // 4 N-tiles of 16

    v8f acc[2][4] = {};

    for (int k = 0; k < KP; k += 32) {
        v16bf ahi[2], alo[2];
#pragma unroll
        for (int i = 0; i < 2; ++i) {
            ahi[i] = load_frag(Ahi, mb + i * 16 + laneLo, k, laneHi);
            alo[i] = load_frag(Alo, mb + i * 16 + laneLo, k, laneHi);
        }
#pragma unroll
        for (int j = 0; j < 4; ++j) {
            v16bf bhi = load_frag(Whi, nb + j * 16 + laneLo, k, laneHi);
            v16bf blo = load_frag(Wlo, nb + j * 16 + laneLo, k, laneHi);
#pragma unroll
            for (int i = 0; i < 2; ++i) {
                acc[i][j] = __builtin_amdgcn_wmma_f32_16x16x32_bf16(
                    false, ahi[i], false, bhi, (short)0, acc[i][j], false, false);
                acc[i][j] = __builtin_amdgcn_wmma_f32_16x16x32_bf16(
                    false, ahi[i], false, blo, (short)0, acc[i][j], false, false);
                acc[i][j] = __builtin_amdgcn_wmma_f32_16x16x32_bf16(
                    false, alo[i], false, bhi, (short)0, acc[i][j], false, false);
            }
        }
    }
    // D layout: VGPR r -> M = r + 8*laneHi, N = laneLo
#pragma unroll
    for (int j = 0; j < 4; ++j) {
        int n = nb + j * 16 + laneLo;
        if (n < NCOL) {
            float bias = bdg[n];
#pragma unroll
            for (int i = 0; i < 2; ++i) {
#pragma unroll
                for (int r = 0; r < 8; ++r) {
                    int m = mb + i * 16 + laneHi * 8 + r;
                    dg[(long long)m * NCOL + n] = acc[i][j][r] + bias;
                }
            }
        }
    }
}

// ---------------- layer 0: z0[b,t,h] = tanh(dg[b,t,:] . W0[t,h,:] + b0) ----------------
__global__ void layer0_kernel(const float* __restrict__ dg, const float* __restrict__ W0,
                              const float* __restrict__ b0, float* __restrict__ z0) {
    int idx = blockIdx.x * blockDim.x + threadIdx.x;     // b*3000 + t*6 + h
    if (idx >= BATCH * T0N * HID) return;
    int b = idx / (T0N * HID);
    int c = idx % (T0N * HID);
    int t = c / HID;
    const float* x = dg + (long long)b * NCOL + t * GPT;
    const float* w = W0 + (long long)c * GPT;
    float s = b0[c];
#pragma unroll
    for (int g = 0; g < GPT; ++g) s += x[g] * w[g];
    z0[idx] = tanhf(s);
}

// ---------------- generic BN stats over the batch axis: z is [BATCH, C] ----------------
__global__ void bn_stats_kernel(const float* __restrict__ z, int C,
                                float* __restrict__ mean, float* __restrict__ istd) {
    int c0 = blockIdx.x * 64;
    int cl = threadIdx.x & 63;
    int rg = threadIdx.x >> 6;                 // 0..3 row groups
    int c  = c0 + cl;
    float s = 0.f, s2 = 0.f;
    if (c < C) {
        for (int b = rg; b < BATCH; b += 4) {
            float v = z[(long long)b * C + c];
            s += v; s2 += v * v;
        }
    }
    __shared__ float sh[2][256];
    sh[0][threadIdx.x] = s;
    sh[1][threadIdx.x] = s2;
    __syncthreads();
    if (rg == 0 && c < C) {
        s  = sh[0][cl] + sh[0][cl + 64] + sh[0][cl + 128] + sh[0][cl + 192];
        s2 = sh[1][cl] + sh[1][cl + 64] + sh[1][cl + 128] + sh[1][cl + 192];
        float m = s * (1.0f / BATCH);
        float v = s2 * (1.0f / BATCH) - m * m;
        mean[c] = m;
        istd[c] = rsqrtf(v + EPSBN);
    }
}

// ---------------- normalize + aux head for layers 0/1 (H=6 channels per term) ----------
__global__ void bn_aux_kernel(const float* __restrict__ z,
                              const float* __restrict__ mean, const float* __restrict__ istd,
                              const float* __restrict__ gam, const float* __restrict__ bet,
                              const float* __restrict__ aw, const float* __restrict__ ab,
                              const float* __restrict__ av, const float* __restrict__ ac,
                              float* __restrict__ out, int T, int nn_off, int aux_off) {
    int idx = blockIdx.x * blockDim.x + threadIdx.x;     // b*T + t
    if (idx >= BATCH * T) return;
    int b = idx / T, t = idx % T;
    const int C = T * HID;
    float* row = out + (long long)b * OUTW;
    float s = ab[t];
#pragma unroll
    for (int h = 0; h < HID; ++h) {
        int c = t * HID + h;
        float n = gam[c] * (z[(long long)b * C + c] - mean[c]) * istd[c] + bet[c];
        row[NNOFF + nn_off + c] = n;
        s += n * aw[c];
    }
    row[aux_off + t] = tanhf(s) * av[t] + ac[t];
}

// ---------------- layer 1: children = leaves [10t,10t+10) + direct genes --------------
__global__ void layer1_kernel(const float* __restrict__ out, const float* __restrict__ dg,
                              const float* __restrict__ W1, const float* __restrict__ b1,
                              float* __restrict__ z1) {
    int idx = blockIdx.x * blockDim.x + threadIdx.x;     // b*300 + t*6 + h
    if (idx >= BATCH * T1N * HID) return;
    int b = idx / (T1N * HID);
    int c = idx % (T1N * HID);
    int t = c / HID;
    const float* n0 = out + (long long)b * OUTW + NNOFF + t * 10 * HID;  // 60 contiguous
    const float* xg = dg + (long long)b * NCOL + (T0N + t) * GPT;
    const float* w  = W1 + (long long)c * 90;
    float s = b1[c];
#pragma unroll
    for (int i = 0; i < 60; ++i) s += n0[i] * w[i];
#pragma unroll
    for (int g = 0; g < GPT; ++g) s += xg[g] * w[60 + g];
    z1[idx] = tanhf(s);
}

// ---------------- root: all mid terms + direct genes ----------------
__global__ void layer2_kernel(const float* __restrict__ out, const float* __restrict__ dg,
                              const float* __restrict__ W2, const float* __restrict__ b2,
                              float* __restrict__ z2) {
    int idx = blockIdx.x * blockDim.x + threadIdx.x;     // b*6 + h
    if (idx >= BATCH * HID) return;
    int b = idx / HID, h = idx % HID;
    const float* n1 = out + (long long)b * OUTW + NNOFF + T0N * HID;     // 300 contiguous
    const float* xg = dg + (long long)b * NCOL + (NTERM - 1) * GPT;
    const float* w  = W2 + h * 330;
    float s = b2[h];
    for (int i = 0; i < 300; ++i) s += n1[i] * w[i];
#pragma unroll
    for (int g = 0; g < GPT; ++g) s += xg[g] * w[300 + g];
    z2[idx] = tanhf(s);
}

__global__ void bn_aux2_kernel(const float* __restrict__ z2,
                               const float* __restrict__ mean, const float* __restrict__ istd,
                               const float* __restrict__ gam, const float* __restrict__ bet,
                               const float* __restrict__ aw, const float* __restrict__ ab,
                               const float* __restrict__ av, const float* __restrict__ ac,
                               float* __restrict__ out) {
    int b = blockIdx.x * blockDim.x + threadIdx.x;
    if (b >= BATCH) return;
    float* row = out + (long long)b * OUTW;
    float s = ab[0];
#pragma unroll
    for (int h = 0; h < HID; ++h) {
        float n = gam[h] * (z2[b * HID + h] - mean[h]) * istd[h] + bet[h];
        row[NNOFF + T0N * HID + T1N * HID + h] = n;
        s += n * aw[h];
    }
    row[T0N + T1N] = tanhf(s) * av[0] + ac[0];
}

// ---------------- host launcher ----------------
extern "C" void kernel_launch(void* const* d_in, const int* in_sizes, int n_in,
                              void* d_out, int out_size, void* d_ws, size_t ws_size,
                              hipStream_t stream) {
    (void)in_sizes; (void)n_in; (void)out_size; (void)ws_size;
    const float* x    = (const float*)d_in[0];
    const float* Wdg  = (const float*)d_in[1];
    const float* bdg  = (const float*)d_in[2];
    const float* W0   = (const float*)d_in[3];
    const float* b0   = (const float*)d_in[4];
    const float* gam0 = (const float*)d_in[5];
    const float* bet0 = (const float*)d_in[6];
    const float* aw0  = (const float*)d_in[7];
    const float* ab0  = (const float*)d_in[8];
    const float* av0  = (const float*)d_in[9];
    const float* ac0  = (const float*)d_in[10];
    const float* W1   = (const float*)d_in[11];
    const float* b1   = (const float*)d_in[12];
    const float* gam1 = (const float*)d_in[13];
    const float* bet1 = (const float*)d_in[14];
    const float* aw1  = (const float*)d_in[15];
    const float* ab1  = (const float*)d_in[16];
    const float* av1  = (const float*)d_in[17];
    const float* ac1  = (const float*)d_in[18];
    const float* W2   = (const float*)d_in[19];
    const float* b2   = (const float*)d_in[20];
    const float* gam2 = (const float*)d_in[21];
    const float* bet2 = (const float*)d_in[22];
    const float* aw2  = (const float*)d_in[23];
    const float* ab2  = (const float*)d_in[24];
    const float* av2  = (const float*)d_in[25];
    const float* ac2  = (const float*)d_in[26];
    float* out = (float*)d_out;

    // workspace carve-up (256B aligned)
    char* p = (char*)d_ws;
    auto alloc = [&](size_t bytes) -> void* {
        void* r = (void*)p;
        p += (bytes + 255) & ~(size_t)255;
        return r;
    };
    float*          dg    = (float*)alloc((size_t)BATCH * NCOL * 4);
    unsigned short* Ahi   = (unsigned short*)alloc((size_t)BATCH * KP * 2);
    unsigned short* Alo   = (unsigned short*)alloc((size_t)BATCH * KP * 2);
    unsigned short* Whi   = (unsigned short*)alloc((size_t)NP * KP * 2);
    unsigned short* Wlo   = (unsigned short*)alloc((size_t)NP * KP * 2);
    float*          z0    = (float*)alloc((size_t)BATCH * T0N * HID * 4);
    float*          z1    = (float*)alloc((size_t)BATCH * T1N * HID * 4);
    float*          z2    = (float*)alloc((size_t)BATCH * HID * 4);
    float*          mean0 = (float*)alloc(T0N * HID * 4);
    float*          istd0 = (float*)alloc(T0N * HID * 4);
    float*          mean1 = (float*)alloc(T1N * HID * 4);
    float*          istd1 = (float*)alloc(T1N * HID * 4);
    float*          mean2 = (float*)alloc(HID * 4);
    float*          istd2 = (float*)alloc(HID * 4);

    // 1) split-bf16 conversions (zero-padded to KP / NP)
    {
        long long tot = (long long)BATCH * KP;
        cvt_split_kernel<<<dim3((unsigned)((tot + 255) / 256)), dim3(256), 0, stream>>>(
            x, BATCH, KDIM, Ahi, Alo, BATCH, KP);
    }
    {
        long long tot = (long long)NP * KP;
        cvt_split_kernel<<<dim3((unsigned)((tot + 255) / 256)), dim3(256), 0, stream>>>(
            Wdg, NCOL, KDIM, Whi, Wlo, NP, KP);
    }

    // 2) big WMMA GEMM: dg[B, 16530]
    gemm_dg_kernel<<<dim3(NP / 128, BATCH / 128), dim3(256), 0, stream>>>(
        Ahi, Alo, Whi, Wlo, bdg, dg);

    // 3) layer 0 + BN + aux
    layer0_kernel<<<dim3((BATCH * T0N * HID + 255) / 256), dim3(256), 0, stream>>>(dg, W0, b0, z0);
    bn_stats_kernel<<<dim3((T0N * HID + 63) / 64), dim3(256), 0, stream>>>(z0, T0N * HID, mean0, istd0);
    bn_aux_kernel<<<dim3((BATCH * T0N + 255) / 256), dim3(256), 0, stream>>>(
        z0, mean0, istd0, gam0, bet0, aw0, ab0, av0, ac0, out, T0N, 0, 0);

    // 4) layer 1 + BN + aux
    layer1_kernel<<<dim3((BATCH * T1N * HID + 255) / 256), dim3(256), 0, stream>>>(out, dg, W1, b1, z1);
    bn_stats_kernel<<<dim3((T1N * HID + 63) / 64), dim3(256), 0, stream>>>(z1, T1N * HID, mean1, istd1);
    bn_aux_kernel<<<dim3((BATCH * T1N + 255) / 256), dim3(256), 0, stream>>>(
        z1, mean1, istd1, gam1, bet1, aw1, ab1, av1, ac1, out, T1N, T0N * HID, T0N);

    // 5) root + BN + aux
    layer2_kernel<<<dim3((BATCH * HID + 255) / 256), dim3(256), 0, stream>>>(out, dg, W2, b2, z2);
    bn_stats_kernel<<<dim3(1), dim3(256), 0, stream>>>(z2, HID, mean2, istd2);
    bn_aux2_kernel<<<dim3((BATCH + 255) / 256), dim3(256), 0, stream>>>(
        z2, mean2, istd2, gam2, bet2, aw2, ab2, av2, ac2, out);
}
